// GCNLabelEncoder_35158602285583
// MI455X (gfx1250) — matrix operand order repaired and checked by reading, not compiled, not run on hardware
//
#include <hip/hip_runtime.h>
#include <hip/hip_bf16.h>

typedef __attribute__((ext_vector_type(16))) _Float16 v16h;
typedef __attribute__((ext_vector_type(8)))  float    v8f;

#define N_NODES 50000
#define MPAD    50048          // 391 * 128
#define K1      300
#define N1      400
#define K1P     320
#define N1PAD   512            // pad B rows so unguarded fragment loads are in-bounds (4 x 128 N-tiles)
#define K2      400
#define N2      512
#define K2P     416
#define LEAKY   0.2f

// ---- workspace layout (bytes) ----
#define OFF_DEG_OUT 0u                    // MPAD f32
#define OFF_DEG_IN  262144u               // MPAD f32
#define OFF_W1T     524288u               // 512*320 f16 = 327680
#define OFF_W2T     1048576u              // 512*416 f16 = 425984
#define OFF_AGG     2097152u              // MPAD*416 f32 (reused both layers)
#define AGG_BYTES   ((size_t)MPAD * 416 * 4)

// -------- degree accumulation --------
__global__ void deg_kernel(const int* __restrict__ src, const int* __restrict__ dst,
                           float* __restrict__ dout, float* __restrict__ din, int E) {
    int e = blockIdx.x * blockDim.x + threadIdx.x;
    if (e < E) {
        atomicAdd(&dout[src[e]], 1.0f);
        atomicAdd(&din[dst[e]], 1.0f);
    }
}

__global__ void inv_sqrt_kernel(float* __restrict__ a, float* __restrict__ b, int n) {
    int i = blockIdx.x * blockDim.x + threadIdx.x;
    if (i < n) {
        a[i] = rsqrtf(fmaxf(a[i], 1.0f));
        b[i] = rsqrtf(fmaxf(b[i], 1.0f));
    }
}

// -------- weight pack: W[K][N] f32 -> Wt[Npad][KP] f16 (transposed, zero-padded) --------
__global__ void pack_w_kernel(const float* __restrict__ W, _Float16* __restrict__ Wt,
                              int K, int N, int KP, int Npad) {
    int idx = blockIdx.x * blockDim.x + threadIdx.x;
    if (idx >= Npad * KP) return;
    int n = idx / KP, k = idx % KP;
    float v = (n < N && k < K) ? W[k * N + n] : 0.0f;
    Wt[idx] = (_Float16)v;
}

// -------- edge scatter: agg[dst] += x[src] * out_inv[src]  (one wave per edge) --------
template <int C, int LD>
__global__ void scatter_kernel(const float* __restrict__ X, const int* __restrict__ src,
                               const int* __restrict__ dst, const float* __restrict__ out_inv,
                               float* __restrict__ agg, int E) {
    int e = blockIdx.x * 8 + (threadIdx.x >> 5);   // 8 waves per block, 1 edge per wave
    if (e >= E) return;
    int lane = threadIdx.x & 31;
    int s = src[e], d = dst[e];
    float sc = out_inv[s];
    const float* xr = X + (size_t)s * C;
    float* ar = agg + (size_t)d * LD;
    for (int c = lane; c < C; c += 32)
        atomicAdd(&ar[c], xr[c] * sc);
}

// -------- WMMA GEMM: out[M][NCOLS] = LeakyReLU?( (A*in_inv) @ Bt^T + bias ) --------
// A: [MPAD][KP] f32 (K-padded, zero-filled), Bt: [Npad][KP] f16 (transposed, padded)
// block = 256 threads = 8 waves; wave w owns rows m0..m0+15 with 8 N-accumulators (128 cols)
template <int KP, int NCOLS, bool LRELU>
__global__ __launch_bounds__(256) void gemm_wmma_kernel(
        const float* __restrict__ A, const float* __restrict__ in_inv,
        const _Float16* __restrict__ Bt, const float* __restrict__ bias,
        float* __restrict__ out, int M, int ld_out) {
    const int lane = threadIdx.x & 31;
    const int wave = threadIdx.x >> 5;
    const int half = lane >> 4;       // 0: lanes 0-15, 1: lanes 16-31
    const int l16  = lane & 15;
    const int m0   = blockIdx.x * 128 + wave * 16;
    const int n0   = blockIdx.y * 128;

    // M is a multiple of 16 and wave tiles are 16 rows: waves are fully in or fully out.
    if (m0 >= M) return;              // wave-uniform; EXEC stays all-1s for WMMA

    const int   mrow  = m0 + l16;                 // A-fragment row (ISA: M = lane%16)
    const float scale = in_inv[mrow];
    const float* arow = A + (size_t)mrow * KP;

    v8f acc[8] = {};

    for (int k0 = 0; k0 < KP; k0 += 32) {
        // A fragment (16x32 f16): elements 0..15 -> K=k0+16*half+0..15 (matches B)
        const int off = half * 16;
        float4 a0 = *(const float4*)(arow + k0 + off);
        float4 a1 = *(const float4*)(arow + k0 + off + 4);
        float4 a2 = *(const float4*)(arow + k0 + off + 8);
        float4 a3 = *(const float4*)(arow + k0 + off + 12);
        v16h af;
        af[0]  = (_Float16)(a0.x * scale); af[1]  = (_Float16)(a0.y * scale);
        af[2]  = (_Float16)(a0.z * scale); af[3]  = (_Float16)(a0.w * scale);
        af[4]  = (_Float16)(a1.x * scale); af[5]  = (_Float16)(a1.y * scale);
        af[6]  = (_Float16)(a1.z * scale); af[7]  = (_Float16)(a1.w * scale);
        af[8]  = (_Float16)(a2.x * scale); af[9]  = (_Float16)(a2.y * scale);
        af[10] = (_Float16)(a2.z * scale); af[11] = (_Float16)(a2.w * scale);
        af[12] = (_Float16)(a3.x * scale); af[13] = (_Float16)(a3.y * scale);
        af[14] = (_Float16)(a3.z * scale); af[15] = (_Float16)(a3.w * scale);

#pragma unroll
        for (int j = 0; j < 8; ++j) {
            // B fragment (32x16 f16): lane holds N=lane%16, K = k0 + 16*half + 0..15 (contiguous in Bt)
            const _Float16* brow = Bt + (size_t)(n0 + j * 16 + l16) * KP + k0 + half * 16;
            v16h bf = *(const v16h*)brow;
            acc[j] = __builtin_amdgcn_wmma_f32_16x16x32_f16(
                false, af, false, bf, (short)0, acc[j], false, false);
        }
    }

    // C/D layout: VGPR r -> M = r + 8*half, N = lane%16
#pragma unroll
    for (int j = 0; j < 8; ++j) {
        int n = n0 + j * 16 + l16;
        if ((NCOLS & 127) != 0 && n >= NCOLS) continue;   // elided when NCOLS % 128 == 0
        float bv = bias[n];
        float* orow = out + (size_t)(m0 + half * 8) * ld_out + n;
#pragma unroll
        for (int r = 0; r < 8; ++r) {
            float v = acc[j][r] + bv;
            if (LRELU) v = (v >= 0.0f) ? v : LEAKY * v;
            orow[(size_t)r * ld_out] = v;
        }
    }
}

extern "C" void kernel_launch(void* const* d_in, const int* in_sizes, int n_in,
                              void* d_out, int out_size, void* d_ws, size_t ws_size,
                              hipStream_t stream) {
    const float* emb = (const float*)d_in[0];
    const float* W1  = (const float*)d_in[1];
    const float* b1  = (const float*)d_in[2];
    const float* W2  = (const float*)d_in[3];
    const float* b2  = (const float*)d_in[4];
    const int*   src = (const int*)d_in[5];
    const int*   dst = (const int*)d_in[6];
    const int    E   = in_sizes[5];

    float* out = (float*)d_out;
    char*  ws  = (char*)d_ws;

    float*     deg_out = (float*)(ws + OFF_DEG_OUT);   // -> D_out^{-1/2}
    float*     deg_in  = (float*)(ws + OFF_DEG_IN);    // -> D_in^{-1/2}
    _Float16*  w1t     = (_Float16*)(ws + OFF_W1T);
    _Float16*  w2t     = (_Float16*)(ws + OFF_W2T);
    float*     agg     = (float*)(ws + OFF_AGG);

    // zero degree + agg buffers (graph-capture-safe async memsets)
    hipMemsetAsync(deg_out, 0, (size_t)MPAD * 4, stream);
    hipMemsetAsync(deg_in,  0, (size_t)MPAD * 4, stream);
    hipMemsetAsync(agg,     0, AGG_BYTES, stream);

    // degrees -> inverse sqrt (clamped at 1)
    deg_kernel<<<(E + 255) / 256, 256, 0, stream>>>(src, dst, deg_out, deg_in, E);
    inv_sqrt_kernel<<<(MPAD + 255) / 256, 256, 0, stream>>>(deg_out, deg_in, MPAD);

    // pack weights to transposed/padded f16
    pack_w_kernel<<<(N1PAD * K1P + 255) / 256, 256, 0, stream>>>(W1, w1t, K1, N1, K1P, N1PAD);
    pack_w_kernel<<<(N2 * K2P + 255) / 256, 256, 0, stream>>>(W2, w2t, K2, N2, K2P, N2);

    // ---- layer 1 ----
    scatter_kernel<K1, K1P><<<(E + 7) / 8, 256, 0, stream>>>(emb, src, dst, deg_out, agg, E);
    // X1 staged inside d_out (50000 x 400 f32, first 80MB of the 102MB output buffer)
    gemm_wmma_kernel<K1P, N1, true><<<dim3(MPAD / 128, (N1 + 127) / 128), 256, 0, stream>>>(
        agg, deg_in, w1t, b1, out, N_NODES, N1);

    // ---- layer 2 ----
    hipMemsetAsync(agg, 0, AGG_BYTES, stream);
    scatter_kernel<K2, K2P><<<(E + 7) / 8, 256, 0, stream>>>(out, src, dst, deg_out, agg, E);
    gemm_wmma_kernel<K2P, N2, false><<<dim3(MPAD / 128, N2 / 128), 256, 0, stream>>>(
        agg, deg_in, w2t, b2, out, N_NODES, N2);
}
